// SimilarityLayer_15032385536515
// MI455X (gfx1250) — matrix operand order repaired and checked by reading, not compiled, not run on hardware
//
#include <hip/hip_runtime.h>

// ---------------------------------------------------------------------------
// SimilarityLayer: out[b,c,q] = dot(H[b,c,:]*w_hu, U[b,q,:]) + s_h[b,c] + s_u[b,q]
// B=64, C=512, Q=64, D=512, all f32.
//
// MI455X strategy:
//  * Split-bf16 GEMM on V_WMMA_F32_16X16X32_BF16: x*y ~= xh*yh + xh*yl + xl*yh
//    (hi = truncated top-16 bits, lo = remainder). Keeps ~f32 accuracy at bf16
//    matrix-core rates -> kernel stays at the ~3.4us HBM roofline (80 MB @
//    23.3 TB/s; whole problem fits in 192 MB L2).
//  * Tiles are converted to packed-bf16 hi/lo ONCE during LDS staging, stored
//    in WMMA fragment order. Fragment build = ds_load_b128 + bit_cast, no VALU.
//    (v1 converted per-wave per-fragment -> 877 VALU/loop; this removes it.)
//  * Pack 2 bf16/dword with one v_perm_b32 (__builtin_amdgcn_perm).
//  * Software pipeline: next K-slice global loads issued into registers right
//    after the consume barrier, overlapping the WMMA work.
//  * s_h/s_u: per-thread register partials during staging (each thread owns
//    fixed rows), one ds_add_f32 each at the end.
// ---------------------------------------------------------------------------

typedef __attribute__((ext_vector_type(16))) __bf16        v16bf;
typedef __attribute__((ext_vector_type(8)))  float         v8f;
typedef __attribute__((ext_vector_type(8)))  unsigned int  v8u;

#define B_SZ   64
#define C_SZ   512
#define Q_SZ   64
#define D_SZ   512
#define KT     32      // K per step (one bf16 WMMA K)
#define ROWS   128     // C-rows per workgroup
#define PK_LD  20      // packed row stride in dwords (16 used + 4 pad; 16B aligned,
                       // 16 rows -> 16 distinct banks)

// pack {bf16(x0), bf16(x1)} (truncating) into one dword with a single v_perm_b32
static __device__ __forceinline__ unsigned int bfpack(float x0, float x1) {
    return __builtin_amdgcn_perm(__float_as_uint(x1), __float_as_uint(x0), 0x07060302u);
}
static __device__ __forceinline__ float bftrunc(float x) {
    return __uint_as_float(__float_as_uint(x) & 0xFFFF0000u);
}

__global__ __launch_bounds__(256)
void similarity_wmma_kernel(const float* __restrict__ H,
                            const float* __restrict__ U,
                            const float* __restrict__ w,
                            float* __restrict__ out)
{
    __shared__ unsigned int lHhi[ROWS * PK_LD];   // H*w_hu, hi bf16 pairs
    __shared__ unsigned int lHlo[ROWS * PK_LD];   // H*w_hu, lo bf16 pairs
    __shared__ unsigned int lUhi[Q_SZ * PK_LD];
    __shared__ unsigned int lUlo[Q_SZ * PK_LD];
    __shared__ float lw[3 * D_SZ];                // w_h | w_u | w_hu
    __shared__ float lsh[ROWS];
    __shared__ float lsu[Q_SZ];

    const int tid  = threadIdx.x;
    const int lane = tid & 31;
    const int wave = tid >> 5;            // 0..7, owns 16 C-rows
    const int hi16 = (lane >> 4) & 1;     // lane half (WMMA operand layout)
    const int mrow = lane & 15;

    const int b  = blockIdx.y;
    const int c0 = blockIdx.x * ROWS;

    const float* Hb = H + ((size_t)b * C_SZ + c0) * D_SZ;
    const float* Ub = U + (size_t)b * Q_SZ * D_SZ;

    // Fixed per-thread staging coordinates (same rows every K-step)
    int hrow[4], hc4[4], urow[2], uc4[2];
    #pragma unroll
    for (int i = 0; i < 4; ++i) {
        int idx = tid + i * 256;          // 1024 float4 = 128x32 H tile
        hrow[i] = idx >> 3;
        hc4[i]  = (idx & 7) << 2;
    }
    #pragma unroll
    for (int i = 0; i < 2; ++i) {
        int idx = tid + i * 256;          // 512 float4 = 64x32 U tile
        urow[i] = idx >> 3;
        uc4[i]  = (idx & 7) << 2;
    }

    // Stage w once (6 KB); zero the rank-1 reduction buffers
    #pragma unroll
    for (int i = 0; i < 6; ++i) lw[tid + i * 256] = w[tid + i * 256];
    if (tid < ROWS)             lsh[tid]        = 0.0f;
    else if (tid < ROWS + Q_SZ) lsu[tid - ROWS] = 0.0f;

    v8f acc[4] = {};                      // 16 rows x 64 cols per wave
    float shp[4] = {0.f, 0.f, 0.f, 0.f};  // per-thread s_h partials
    float sup[2] = {0.f, 0.f};            // per-thread s_u partials

    // --- pipeline prologue: fetch first K-slice into registers ---
    float4 hreg[4], ureg[2];
    #pragma unroll
    for (int i = 0; i < 4; ++i)
        hreg[i] = *(const float4*)(Hb + (size_t)hrow[i] * D_SZ + hc4[i]);
    #pragma unroll
    for (int i = 0; i < 2; ++i)
        ureg[i] = *(const float4*)(Ub + (size_t)urow[i] * D_SZ + uc4[i]);

    for (int k0 = 0; k0 < D_SZ; k0 += KT) {
        __syncthreads();                  // previous tiles fully consumed

        // --- convert registers -> packed split-bf16 LDS tiles -------------
        #pragma unroll
        for (int i = 0; i < 4; ++i) {
            const float4 v = hreg[i];
            const int kk = k0 + hc4[i];
            shp[i] += v.x * lw[kk] + v.y * lw[kk + 1]
                    + v.z * lw[kk + 2] + v.w * lw[kk + 3];
            float m0 = v.x * lw[2 * D_SZ + kk];
            float m1 = v.y * lw[2 * D_SZ + kk + 1];
            float m2 = v.z * lw[2 * D_SZ + kk + 2];
            float m3 = v.w * lw[2 * D_SZ + kk + 3];
            unsigned int h01 = bfpack(m0, m1), h23 = bfpack(m2, m3);
            unsigned int l01 = bfpack(m0 - bftrunc(m0), m1 - bftrunc(m1));
            unsigned int l23 = bfpack(m2 - bftrunc(m2), m3 - bftrunc(m3));
            const int base = hrow[i] * PK_LD + (hc4[i] >> 1);
            *(uint2*)&lHhi[base] = make_uint2(h01, h23);
            *(uint2*)&lHlo[base] = make_uint2(l01, l23);
        }
        #pragma unroll
        for (int i = 0; i < 2; ++i) {
            const float4 v = ureg[i];
            const int kk = k0 + uc4[i];
            sup[i] += v.x * lw[D_SZ + kk]     + v.y * lw[D_SZ + kk + 1]
                    + v.z * lw[D_SZ + kk + 2] + v.w * lw[D_SZ + kk + 3];
            unsigned int h01 = bfpack(v.x, v.y), h23 = bfpack(v.z, v.w);
            unsigned int l01 = bfpack(v.x - bftrunc(v.x), v.y - bftrunc(v.y));
            unsigned int l23 = bfpack(v.z - bftrunc(v.z), v.w - bftrunc(v.w));
            const int base = urow[i] * PK_LD + (uc4[i] >> 1);
            *(uint2*)&lUhi[base] = make_uint2(h01, h23);
            *(uint2*)&lUlo[base] = make_uint2(l01, l23);
        }

        __syncthreads();                  // tiles ready

        // --- issue next K-slice loads now; latency hidden by WMMA work ----
        if (k0 + KT < D_SZ) {
            const int kn = k0 + KT;
            #pragma unroll
            for (int i = 0; i < 4; ++i)
                hreg[i] = *(const float4*)(Hb + (size_t)hrow[i] * D_SZ + kn + hc4[i]);
            #pragma unroll
            for (int i = 0; i < 2; ++i)
                ureg[i] = *(const float4*)(Ub + (size_t)urow[i] * D_SZ + kn + uc4[i]);
        }

        // --- A fragment: two b128 loads per hi/lo, bit_cast, no VALU ------
        const int abase = (wave * 16 + mrow) * PK_LD + (hi16 << 2);
        v8u a_h, a_l;
        #pragma unroll
        for (int d = 0; d < 4; ++d) {
            a_h[d]     = lHhi[abase + d];
            a_h[d + 4] = lHhi[abase + 8 + d];
            a_l[d]     = lHlo[abase + d];
            a_l[d + 4] = lHlo[abase + 8 + d];
        }
        const v16bf ahi = __builtin_bit_cast(v16bf, a_h);
        const v16bf alo = __builtin_bit_cast(v16bf, a_l);

        // --- 4 column subtiles x 3 WMMAs (hi*hi + hi*lo + lo*hi) ----------
        #pragma unroll
        for (int n = 0; n < 4; ++n) {
            const int bbase = (n * 16 + mrow) * PK_LD + (hi16 << 2);
            v8u b_h, b_l;
            #pragma unroll
            for (int d = 0; d < 4; ++d) {
                b_h[d]     = lUhi[bbase + d];
                b_h[d + 4] = lUhi[bbase + 8 + d];
                b_l[d]     = lUlo[bbase + d];
                b_l[d + 4] = lUlo[bbase + 8 + d];
            }
            const v16bf bhi = __builtin_bit_cast(v16bf, b_h);
            const v16bf blo = __builtin_bit_cast(v16bf, b_l);

            acc[n] = __builtin_amdgcn_wmma_f32_16x16x32_bf16(
                         false, ahi, false, bhi, (short)0, acc[n], false, false);
            acc[n] = __builtin_amdgcn_wmma_f32_16x16x32_bf16(
                         false, ahi, false, blo, (short)0, acc[n], false, false);
            acc[n] = __builtin_amdgcn_wmma_f32_16x16x32_bf16(
                         false, alo, false, bhi, (short)0, acc[n], false, false);
        }
    }

    // --- reduce rank-1 partials (8 threads per H row, 8 per U row) --------
    __syncthreads();
    #pragma unroll
    for (int i = 0; i < 4; ++i) atomicAdd(&lsh[hrow[i]], shp[i]);
    #pragma unroll
    for (int i = 0; i < 2; ++i) atomicAdd(&lsu[urow[i]], sup[i]);
    __syncthreads();

    // --- epilogue: C/D layout: VGPR r -> M=r (lanes<16) / M=8+r (lanes>=16)
    float* outb = out + ((size_t)b * C_SZ + c0) * Q_SZ;
    #pragma unroll
    for (int n = 0; n < 4; ++n) {
        #pragma unroll
        for (int r = 0; r < 8; ++r) {
            int rowin = wave * 16 + (hi16 << 3) + r;
            int q     = n * 16 + mrow;
            outb[(size_t)rowin * Q_SZ + q] = acc[n][r] + lsh[rowin] + lsu[q];
        }
    }
}

extern "C" void kernel_launch(void* const* d_in, const int* in_sizes, int n_in,
                              void* d_out, int out_size, void* d_ws, size_t ws_size,
                              hipStream_t stream) {
    (void)in_sizes; (void)n_in; (void)out_size; (void)d_ws; (void)ws_size;
    const float* H = (const float*)d_in[0];
    const float* U = (const float*)d_in[1];
    const float* w = (const float*)d_in[2];
    float* out = (float*)d_out;

    dim3 grid(C_SZ / ROWS, B_SZ);   // (4, 64) workgroups
    dim3 block(256);                // 8 wave32
    similarity_wmma_kernel<<<grid, block, 0, stream>>>(H, U, w, out);
}
